// FourierAttention_78692390797467
// MI455X (gfx1250) — compile-verified, matrix operand order checked by reading, not run on hardware
//
#include <hip/hip_runtime.h>
#include <hip/hip_bf16.h>

typedef __attribute__((ext_vector_type(16))) _Float16 v16h;
typedef __attribute__((ext_vector_type(8)))  float    v8f;

#define Bb     8
#define Cc     384
#define Nn     1024
#define NHEAD  8
#define HD     48
#define HDP    64     // padded head dim
#define OPAD   512    // NHEAD*HDP padded projection output
#define EPSc   1e-5f

// K-index mapping for 16-bit A/B fragments of v_wmma_*_16x16x32_* (wave32):
// VGPR pair i (0..7), lane-half h: K = (i&3)*2 + (i>>2)*16 + h*8 (+0/+1 within pair)
// -> per lane, a fragment is two contiguous 8-half (16B) runs: vectorizes to b128.
__device__ __forceinline__ int kmap(int i, int half) {
  return ((i & 3) << 1) + ((i >> 2) << 4) + (half << 3);
}

// ---------------------------------------------------------------------------
// Kernel 1: per-(b,c) spectral filter: rfft2 -> *W -> irfft2 (ortho),
// via separable 32-point DFTs with shared cos/sin tables. Writes xr[b,c,h,w].
// ---------------------------------------------------------------------------
__global__ __launch_bounds__(256) void fa_fourier(const float* __restrict__ x,
                                                  const float* __restrict__ cw,
                                                  float* __restrict__ xr) {
  const int bc = blockIdx.x;          // b*Cc + c
  const int b = bc / Cc, c = bc % Cc;
  __shared__ float ct[32], st[32];
  __shared__ float X[32][32];
  __shared__ float Sr[32][17], Si[32][17];
  __shared__ float Yr[32][17], Yi[32][17];
  const int t = threadIdx.x;
  if (t < 32) {
    float ang = 6.28318530718f * (float)t / 32.0f;
    ct[t] = __cosf(ang);
    st[t] = __sinf(ang);
  }
  for (int i = t; i < 1024; i += 256) {
    X[i >> 5][i & 31] = x[(size_t)(b * Nn + i) * Cc + c];
  }
  __syncthreads();
  // stage 1: rfft along W (unnormalized; fold 1/1024 at the end)
  for (int i = t; i < 32 * 17; i += 256) {
    int h = i / 17, kw = i % 17;
    float re = 0.f, im = 0.f;
    for (int w = 0; w < 32; ++w) {
      float v = X[h][w];
      int idx = (w * kw) & 31;
      re += v * ct[idx];
      im -= v * st[idx];
    }
    Sr[h][kw] = re; Si[h][kw] = im;
  }
  __syncthreads();
  // stage 2: full FFT along H + complex weight multiply
  for (int i = t; i < 32 * 17; i += 256) {
    int kh = i / 17, kw = i % 17;
    float re = 0.f, im = 0.f;
    for (int h = 0; h < 32; ++h) {
      int idx = (h * kh) & 31;
      float cr = ct[idx], ci = -st[idx];
      float ar = Sr[h][kw], ai = Si[h][kw];
      re += ar * cr - ai * ci;
      im += ar * ci + ai * cr;
    }
    float wr = cw[((size_t)(c * 32 + kh) * 17 + kw) * 2 + 0];
    float wi = cw[((size_t)(c * 32 + kh) * 17 + kw) * 2 + 1];
    Yr[kh][kw] = re * wr - im * wi;
    Yi[kh][kw] = re * wi + im * wr;
  }
  __syncthreads();
  // stage 3: inverse FFT along H
  for (int i = t; i < 32 * 17; i += 256) {
    int h = i / 17, kw = i % 17;
    float re = 0.f, im = 0.f;
    for (int kh = 0; kh < 32; ++kh) {
      int idx = (h * kh) & 31;
      float cr = ct[idx], ci = st[idx];
      float ar = Yr[kh][kw], ai = Yi[kh][kw];
      re += ar * cr - ai * ci;
      im += ar * ci + ai * cr;
    }
    Sr[h][kw] = re; Si[h][kw] = im;
  }
  __syncthreads();
  // stage 4: irfft along W using Hermitian symmetry
  for (int i = t; i < 1024; i += 256) {
    int h = i >> 5, w = i & 31;
    float acc = Sr[h][0];
    #pragma unroll
    for (int kw = 1; kw < 16; ++kw) {
      int idx = (w * kw) & 31;
      acc += 2.0f * (Sr[h][kw] * ct[idx] - Si[h][kw] * st[idx]);
    }
    {
      int idx = (w * 16) & 31;
      acc += Sr[h][16] * ct[idx] - Si[h][16] * st[idx];
    }
    xr[((size_t)bc << 10) + i] = acc * (1.0f / 1024.0f);
  }
}

// ---------------------------------------------------------------------------
// Kernel 2: depthwise 3x3 conv (pad 1) + inference BN, for q/k/v.
// One block (1024 threads) per (b,c) plane; outputs f16 in (B,N,C).
// ---------------------------------------------------------------------------
__global__ __launch_bounds__(1024) void fa_dwconv_bn(
    const float* __restrict__ xr,
    const float* __restrict__ dwq, const float* __restrict__ dwk, const float* __restrict__ dwv,
    const float* __restrict__ gq, const float* __restrict__ bq,
    const float* __restrict__ mq, const float* __restrict__ vq,
    const float* __restrict__ gk, const float* __restrict__ bk,
    const float* __restrict__ mk, const float* __restrict__ vk,
    const float* __restrict__ gv, const float* __restrict__ bv,
    const float* __restrict__ mv, const float* __restrict__ vv,
    _Float16* __restrict__ qc, _Float16* __restrict__ kc, _Float16* __restrict__ vc) {
  const int bc = blockIdx.x;
  const int b = bc / Cc, c = bc % Cc;
  __shared__ float P[34][34];
  const int t = threadIdx.x;
  const int h = t >> 5, w = t & 31;
  for (int i = t; i < 34 * 34; i += 1024) P[i / 34][i % 34] = 0.0f;
  __syncthreads();
  P[h + 1][w + 1] = xr[((size_t)bc << 10) + t];
  __syncthreads();

  float wq9[9], wk9[9], wv9[9];
  #pragma unroll
  for (int i = 0; i < 9; ++i) {
    wq9[i] = dwq[c * 9 + i];
    wk9[i] = dwk[c * 9 + i];
    wv9[i] = dwv[c * 9 + i];
  }
  float aq = 0.f, ak = 0.f, av = 0.f;
  #pragma unroll
  for (int dy = 0; dy < 3; ++dy)
    #pragma unroll
    for (int dx = 0; dx < 3; ++dx) {
      float p = P[h + dy][w + dx];
      aq += p * wq9[dy * 3 + dx];
      ak += p * wk9[dy * 3 + dx];
      av += p * wv9[dy * 3 + dx];
    }
  float sq = gq[c] * __frsqrt_rn(vq[c] + EPSc), fq = bq[c] - mq[c] * sq;
  float sk = gk[c] * __frsqrt_rn(vk[c] + EPSc), fk = bk[c] - mk[c] * sk;
  float sv = gv[c] * __frsqrt_rn(vv[c] + EPSc), fv = bv[c] - mv[c] * sv;
  size_t o = (size_t)(b * Nn + t) * Cc + c;
  qc[o] = (_Float16)(aq * sq + fq);
  kc[o] = (_Float16)(ak * sk + fk);
  vc[o] = (_Float16)(av * sv + fv);
}

// ---------------------------------------------------------------------------
// Kernel 3: convert wq/wk/wv -> f16 padded (OPAD x Cc) with zero rows for
// head-dim 48..63, and wo -> f16 (Cc x Cc).
// ---------------------------------------------------------------------------
__global__ __launch_bounds__(256) void fa_wcvt(
    const float* __restrict__ wq, const float* __restrict__ wk,
    const float* __restrict__ wv, const float* __restrict__ wo,
    _Float16* __restrict__ wqp, _Float16* __restrict__ wkp,
    _Float16* __restrict__ wvp, _Float16* __restrict__ wof) {
  const int idx = blockIdx.x * 256 + threadIdx.x;
  const int npad = 3 * OPAD * Cc;
  if (idx < npad) {
    int sel = idx / (OPAD * Cc);
    int rem = idx % (OPAD * Cc);
    int op = rem / Cc, k = rem % Cc;
    int head = op >> 6, d = op & 63;
    const float* src = (sel == 0) ? wq : (sel == 1) ? wk : wv;
    _Float16* dst = (sel == 0) ? wqp : (sel == 1) ? wkp : wvp;
    float val = (d < HD) ? src[(size_t)(head * HD + d) * Cc + k] : 0.0f;
    dst[rem] = (_Float16)val;
  } else if (idx < npad + Cc * Cc) {
    int i2 = idx - npad;
    wof[i2] = (_Float16)wo[i2];
  }
}

// ---------------------------------------------------------------------------
// Kernel 4: QKV projections.  Per wave: one 16x16 tile of (8192 x 512padded),
// K=384 in 12 WMMA f16 steps.
//   Q, K stored f16 (B, NHEAD, N, HDP)        (row = token, contiguous head dim)
//   V stored TRANSPOSED f16 (B, NHEAD, HDP, N) (row = head dim, contiguous seq)
// so the attention PV B-fragment reads become contiguous b128 loads.
// ---------------------------------------------------------------------------
__global__ __launch_bounds__(256) void fa_qkv_gemm(
    const _Float16* __restrict__ qc, const _Float16* __restrict__ kc,
    const _Float16* __restrict__ vc,
    const _Float16* __restrict__ wqp, const _Float16* __restrict__ wkp,
    const _Float16* __restrict__ wvp,
    _Float16* __restrict__ Qp, _Float16* __restrict__ Kp, _Float16* __restrict__ Vt) {
  const int wid = blockIdx.x * 8 + (threadIdx.x >> 5);
  const int TILES = 512 * (OPAD / 16);            // 16384 tiles per projection
  const int proj = wid / TILES;
  const int tid = wid % TILES;
  const int tn = tid >> 9;                        // 0..31
  const int tm = tid & 511;                       // 0..511
  const int lane = threadIdx.x & 31;
  const int half = lane >> 4, lr = lane & 15;

  const _Float16* src = (proj == 0) ? qc : (proj == 1) ? kc : vc;
  const _Float16* wgt = (proj == 0) ? wqp : (proj == 1) ? wkp : wvp;

  const _Float16* arow = src + (size_t)(tm * 16 + lr) * Cc;   // activation row M
  const _Float16* bcol = wgt + (size_t)(tn * 16 + lr) * Cc;   // weight row = B column N
  v8f acc = {};
  for (int k0 = 0; k0 < Cc; k0 += 32) {
    v16h a, bf;
    #pragma unroll
    for (int i = 0; i < 8; ++i) {
      int kb = k0 + kmap(i, half);
      a[2 * i]     = arow[kb];
      a[2 * i + 1] = arow[kb + 1];
      bf[2 * i]     = bcol[kb];
      bf[2 * i + 1] = bcol[kb + 1];
    }
    acc = __builtin_amdgcn_wmma_f32_16x16x32_f16(false, a, false, bf,
                                                 (short)0, acc, false, false);
  }
  const int op = tn * 16 + lr;                    // padded output channel
  const int head = op >> 6, d = op & 63;
  if (proj == 2) {
    // V transposed: Vt[(bh*HDP + d)*Nn + n]
    #pragma unroll
    for (int r = 0; r < 8; ++r) {
      int m = tm * 16 + r + 8 * half;
      int bi = m >> 10, n = m & 1023;
      Vt[(((size_t)(bi * NHEAD + head) * HDP + d) << 10) + n] = (_Float16)acc[r];
    }
  } else {
    _Float16* dst = (proj == 0) ? Qp : Kp;
    #pragma unroll
    for (int r = 0; r < 8; ++r) {
      int m = tm * 16 + r + 8 * half;             // global row (b*1024+n)
      int bi = m >> 10, n = m & 1023;
      dst[(((size_t)(bi * NHEAD + head) << 10) + n) * HDP + d] = (_Float16)acc[r];
    }
  }
}

// ---------------------------------------------------------------------------
// Kernel 5: attention with online softmax.  Block = 8 waves, each wave owns
// 16 rows of one (b,head).  Grid = (B*NHEAD) * (N/128) = 512 blocks.
// All four fragment streams (Q, K, P, Vt) are contiguous-pair layouts.
// ---------------------------------------------------------------------------
__global__ __launch_bounds__(256) void fa_attn(
    const _Float16* __restrict__ Qp, const _Float16* __restrict__ Kp,
    const _Float16* __restrict__ Vt, _Float16* __restrict__ Obuf) {
  const int blk = blockIdx.x;
  const int bh = blk >> 3;                        // b*NHEAD + head
  const int mblk = blk & 7;
  const int wave = threadIdx.x >> 5;
  const int lane = threadIdx.x & 31;
  const int half = lane >> 4, lr = lane & 15;
  const int row0 = mblk * 128 + wave * 16;

  const _Float16* Q = Qp + ((size_t)bh << 10) * HDP;
  const _Float16* K = Kp + ((size_t)bh << 10) * HDP;
  const _Float16* Vb = Vt + ((size_t)bh * HDP << 10);

  // Q fragments: 2 chunks of K=32 across head dim 64, held in registers
  v16h qf[2];
  const _Float16* qrow = Q + (size_t)(row0 + lr) * HDP;
  #pragma unroll
  for (int ch = 0; ch < 2; ++ch)
    #pragma unroll
    for (int i = 0; i < 8; ++i) {
      int kb = ch * 32 + kmap(i, half);
      qf[ch][2 * i]     = qrow[kb];
      qf[ch][2 * i + 1] = qrow[kb + 1];
    }

  v8f zacc = {};
  v8f oacc[4];
  oacc[0] = zacc; oacc[1] = zacc; oacc[2] = zacc; oacc[3] = zacc;
  float mrow[8], lrow[8];
  #pragma unroll
  for (int r = 0; r < 8; ++r) { mrow[r] = -1e30f; lrow[r] = 0.0f; }
  const float scale = 0.05103103630798288f;       // 384^-0.5

  __shared__ _Float16 pbuf[8][16 * 32];           // per-wave P bounce buffer
  _Float16* myp = pbuf[wave];

  for (int j0 = 0; j0 < Nn; j0 += 32) {
    // --- S = Q K^T for two 16-col tiles ---
    v8f s[2];
    #pragma unroll
    for (int ctile = 0; ctile < 2; ++ctile) {
      v8f sacc = zacc;
      const _Float16* krow = K + (size_t)(j0 + ctile * 16 + lr) * HDP;
      #pragma unroll
      for (int ch = 0; ch < 2; ++ch) {
        v16h bf;
        #pragma unroll
        for (int i = 0; i < 8; ++i) {
          int kb = ch * 32 + kmap(i, half);
          bf[2 * i]     = krow[kb];
          bf[2 * i + 1] = krow[kb + 1];
        }
        sacc = __builtin_amdgcn_wmma_f32_16x16x32_f16(false, qf[ch], false, bf,
                                                      (short)0, sacc, false, false);
      }
      s[ctile] = sacc;
    }
    // --- online softmax update (row M = r + 8*half lives in one half-wave) ---
    float p0v[8], p1v[8];
    #pragma unroll
    for (int r = 0; r < 8; ++r) {
      float v0 = s[0][r] * scale, v1 = s[1][r] * scale;
      float mx = fmaxf(v0, v1);
      #pragma unroll
      for (int dlt = 1; dlt < 16; dlt <<= 1) mx = fmaxf(mx, __shfl_xor(mx, dlt, 32));
      float mn = fmaxf(mrow[r], mx);
      float corr = __expf(mrow[r] - mn);
      float p0 = __expf(v0 - mn), p1 = __expf(v1 - mn);
      float ps = p0 + p1;
      #pragma unroll
      for (int dlt = 1; dlt < 16; dlt <<= 1) ps += __shfl_xor(ps, dlt, 32);
      lrow[r] = lrow[r] * corr + ps;
      mrow[r] = mn;
      #pragma unroll
      for (int nt = 0; nt < 4; ++nt) oacc[nt][r] *= corr;
      p0v[r] = p0; p1v[r] = p1;
    }
    // --- D-layout -> A-layout for P via LDS bounce ---
    #pragma unroll
    for (int r = 0; r < 8; ++r) {
      myp[(r + 8 * half) * 32 + lr]      = (_Float16)p0v[r];
      myp[(r + 8 * half) * 32 + 16 + lr] = (_Float16)p1v[r];
    }
    __syncthreads();
    v16h pf;
    const _Float16* prow = myp + lr * 32;
    #pragma unroll
    for (int i = 0; i < 8; ++i) {
      int kb = kmap(i, half);
      pf[2 * i]     = prow[kb];
      pf[2 * i + 1] = prow[kb + 1];
    }
    // --- O += P V : Vt row = head-dim, contiguous along seq -> b128 loads ---
    #pragma unroll
    for (int nt = 0; nt < 4; ++nt) {
      v16h vf;
      int nn = nt * 16 + lr;
      const _Float16* vrow = Vb + ((size_t)nn << 10) + j0;
      #pragma unroll
      for (int i = 0; i < 8; ++i) {
        int kb = kmap(i, half);
        vf[2 * i]     = vrow[kb];
        vf[2 * i + 1] = vrow[kb + 1];
      }
      oacc[nt] = __builtin_amdgcn_wmma_f32_16x16x32_f16(false, pf, false, vf,
                                                        (short)0, oacc[nt], false, false);
    }
  }
  // --- finalize: divide by l, drop padded dims, write f16 (B,N,C) ---
  const int bi = bh >> 3, head = bh & 7;
  #pragma unroll
  for (int r = 0; r < 8; ++r) {
    float inv = 1.0f / lrow[r];
    int m = row0 + r + 8 * half;
    #pragma unroll
    for (int nt = 0; nt < 4; ++nt) {
      int d = nt * 16 + lr;
      if (d < HD) {
        Obuf[((size_t)(bi * Nn + m)) * Cc + head * HD + d] = (_Float16)(oacc[nt][r] * inv);
      }
    }
  }
}

// ---------------------------------------------------------------------------
// Kernel 6: output projection + bias + spectral residual.  Per wave one
// 16x16 tile of (8192 x 384); residual read transposed from xr.
// ---------------------------------------------------------------------------
__global__ __launch_bounds__(256) void fa_oproj(
    const _Float16* __restrict__ Obuf, const _Float16* __restrict__ wof,
    const float* __restrict__ bo, const float* __restrict__ xr,
    float* __restrict__ out) {
  const int wid = blockIdx.x * 8 + (threadIdx.x >> 5);
  const int tn = wid % 24;                        // 384/16
  const int tm = wid / 24;                        // 0..511
  const int lane = threadIdx.x & 31;
  const int half = lane >> 4, lr = lane & 15;

  const _Float16* arow = Obuf + (size_t)(tm * 16 + lr) * Cc;
  const _Float16* bcol = wof + (size_t)(tn * 16 + lr) * Cc;  // wo row n = B column
  v8f acc = {};
  for (int k0 = 0; k0 < Cc; k0 += 32) {
    v16h a, bf;
    #pragma unroll
    for (int i = 0; i < 8; ++i) {
      int kb = k0 + kmap(i, half);
      a[2 * i]     = arow[kb];
      a[2 * i + 1] = arow[kb + 1];
      bf[2 * i]     = bcol[kb];
      bf[2 * i + 1] = bcol[kb + 1];
    }
    acc = __builtin_amdgcn_wmma_f32_16x16x32_f16(false, a, false, bf,
                                                 (short)0, acc, false, false);
  }
  const int cn = tn * 16 + lr;
  const float bias = bo[cn];
  #pragma unroll
  for (int r = 0; r < 8; ++r) {
    int m = tm * 16 + r + 8 * half;
    int bi = m >> 10, npos = m & 1023;
    float res = xr[(((size_t)(bi * Cc + cn)) << 10) + npos];  // x_fft residual
    out[(size_t)m * Cc + cn] = acc[r] + bias + res;
  }
}

// ---------------------------------------------------------------------------
extern "C" void kernel_launch(void* const* d_in, const int* in_sizes, int n_in,
                              void* d_out, int out_size, void* d_ws, size_t ws_size,
                              hipStream_t stream) {
  (void)in_sizes; (void)n_in; (void)out_size; (void)ws_size;
  const float* x   = (const float*)d_in[0];
  const float* cw  = (const float*)d_in[1];
  const float* dwq = (const float*)d_in[2];
  const float* dwk = (const float*)d_in[3];
  const float* dwv = (const float*)d_in[4];
  const float* gq = (const float*)d_in[5],  *bq = (const float*)d_in[6];
  const float* mq = (const float*)d_in[7],  *vq = (const float*)d_in[8];
  const float* gk = (const float*)d_in[9],  *bk = (const float*)d_in[10];
  const float* mk = (const float*)d_in[11], *vk = (const float*)d_in[12];
  const float* gv = (const float*)d_in[13], *bv = (const float*)d_in[14];
  const float* mv = (const float*)d_in[15], *vv = (const float*)d_in[16];
  const float* wq = (const float*)d_in[17];
  const float* wk = (const float*)d_in[18];
  const float* wv = (const float*)d_in[19];
  const float* wo = (const float*)d_in[20];
  const float* bo = (const float*)d_in[21];
  float* out = (float*)d_out;

  // workspace carve-up (all sizes multiples of 256B)
  char* ws = (char*)d_ws;
  size_t off = 0;
  float* xr = (float*)(ws + off);          off += (size_t)Bb * Cc * Nn * 4;        // 12.6 MB
  _Float16* qc = (_Float16*)(ws + off);    off += (size_t)Bb * Nn * Cc * 2;
  _Float16* kc = (_Float16*)(ws + off);    off += (size_t)Bb * Nn * Cc * 2;
  _Float16* vc = (_Float16*)(ws + off);    off += (size_t)Bb * Nn * Cc * 2;
  _Float16* wqp = (_Float16*)(ws + off);   off += (size_t)OPAD * Cc * 2;
  _Float16* wkp = (_Float16*)(ws + off);   off += (size_t)OPAD * Cc * 2;
  _Float16* wvp = (_Float16*)(ws + off);   off += (size_t)OPAD * Cc * 2;
  _Float16* wof = (_Float16*)(ws + off);   off += (size_t)Cc * Cc * 2;
  _Float16* Qp = (_Float16*)(ws + off);    off += (size_t)Bb * NHEAD * Nn * HDP * 2;
  _Float16* Kp = (_Float16*)(ws + off);    off += (size_t)Bb * NHEAD * Nn * HDP * 2;
  _Float16* Vt = (_Float16*)(ws + off);    off += (size_t)Bb * NHEAD * Nn * HDP * 2;
  _Float16* Obuf = (_Float16*)(ws + off);  off += (size_t)Bb * Nn * Cc * 2;

  // 1. spectral filter
  fa_fourier<<<Bb * Cc, 256, 0, stream>>>(x, cw, xr);
  // 2. depthwise conv + BN
  fa_dwconv_bn<<<Bb * Cc, 1024, 0, stream>>>(xr, dwq, dwk, dwv,
      gq, bq, mq, vq, gk, bk, mk, vk, gv, bv, mv, vv, qc, kc, vc);
  // 3. weight conversion / padding
  {
    int total = 3 * OPAD * Cc + Cc * Cc;
    fa_wcvt<<<(total + 255) / 256, 256, 0, stream>>>(wq, wk, wv, wo,
                                                     wqp, wkp, wvp, wof);
  }
  // 4. QKV projections (WMMA); V stored transposed
  fa_qkv_gemm<<<3 * 512 * (OPAD / 16) / 8, 256, 0, stream>>>(
      qc, kc, vc, wqp, wkp, wvp, Qp, Kp, Vt);
  // 5. attention (WMMA + online softmax)
  fa_attn<<<Bb * NHEAD * (Nn / 128), 256, 0, stream>>>(Qp, Kp, Vt, Obuf);
  // 6. output projection + residual (WMMA)
  fa_oproj<<<512 * 24 / 8, 256, 0, stream>>>(Obuf, wof, bo, xr, out);
}
